// GATv2Encoder_55920474194403
// MI455X (gfx1250) — matrix operand order repaired and checked by reading, not compiled, not run on hardware
//
#include <hip/hip_runtime.h>
#include <hip/hip_bf16.h>
#include <stdint.h>

// ---------------------------------------------------------------------------
// Types for CDNA5 WMMA (wave32): v_wmma_f32_16x16x32_bf16
// ---------------------------------------------------------------------------
typedef __bf16 bf16;
typedef __bf16 v16bf __attribute__((ext_vector_type(16)));
typedef __bf16 v8bf  __attribute__((ext_vector_type(8)));
typedef float  v8f   __attribute__((ext_vector_type(8)));

#define NEG_SLOPE 0.2f

// Order-preserving float<->uint encoding for atomic segment-max
__device__ __forceinline__ unsigned int enc_f(float f) {
    unsigned int u = __float_as_uint(f);
    return (u & 0x80000000u) ? ~u : (u | 0x80000000u);
}
__device__ __forceinline__ float dec_f(unsigned int u) {
    return (u & 0x80000000u) ? __uint_as_float(u ^ 0x80000000u)
                             : __uint_as_float(~u);
}
#define ENC_NEG_INF 0x007FFFFFu  // enc(-inf)

// ---------------------------------------------------------------------------
// Split f32 -> (bf16 hi, bf16 lo) so 3x WMMA recovers ~f32 accuracy
// ---------------------------------------------------------------------------
__global__ void split_bf16_kernel(const float* __restrict__ X,
                                  bf16* __restrict__ hi, bf16* __restrict__ lo,
                                  size_t n) {
    size_t i = (size_t)blockIdx.x * blockDim.x + threadIdx.x;
    if (i >= n) return;
    float v = X[i];
    bf16 h = (bf16)v;
    hi[i] = h;
    lo[i] = (bf16)(v - (float)h);
}

// Weights: convert + transpose so K is contiguous for the B fragment
__global__ void split_w_t_kernel(const float* __restrict__ W,  // [K, Nn] row-major
                                 bf16* __restrict__ hi, bf16* __restrict__ lo,
                                 int K, int Nn) {
    int idx = blockIdx.x * blockDim.x + threadIdx.x;
    if (idx >= K * Nn) return;
    int k = idx / Nn, n2 = idx - k * Nn;
    float v = W[idx];
    bf16 h = (bf16)v;
    size_t t = (size_t)n2 * K + k;  // [Nn, K]
    hi[t] = h;
    lo[t] = (bf16)(v - (float)h);
}

// ---------------------------------------------------------------------------
// WMMA GEMM: C[M,Nn] = A[M,K] * B[K,Nn], A row-major (hi+lo), B stored [Nn,K]
// One wave per 16x16 output tile; waves within a block tile the N dimension.
// 3 wmma per K-step (split-bf16), K/32 steps.
// ---------------------------------------------------------------------------
__global__ void wmma_gemm_bf16x3(const bf16* __restrict__ Ahi, const bf16* __restrict__ Alo,
                                 const bf16* __restrict__ Bhi, const bf16* __restrict__ Blo,
                                 float* __restrict__ C, int M, int K, int Nn) {
    const int lane = threadIdx.x & 31;
    const int wave = threadIdx.x >> 5;
    const int m0 = blockIdx.x * 16;
    const int n0 = wave * 16;
    if (m0 >= M || n0 >= Nn) return;

    const int half = lane >> 4;   // 0: lanes 0-15, 1: lanes 16-31
    const int l16  = lane & 15;
    const size_t arow = (size_t)(m0 + l16) * K;
    const size_t bcol = (size_t)(n0 + l16) * K;

    v8f c = {};
    for (int kk = 0; kk < K; kk += 32) {
        // A fragment: elems 0..7 -> K = kk+8*half+j ; elems 8..15 -> +16
        const int ka = kk + half * 8;
        v8bf a0h = *(const v8bf*)(Ahi + arow + ka);
        v8bf a1h = *(const v8bf*)(Ahi + arow + ka + 16);
        v8bf a0l = *(const v8bf*)(Alo + arow + ka);
        v8bf a1l = *(const v8bf*)(Alo + arow + ka + 16);
        v16bf ahi = __builtin_shufflevector(a0h, a1h, 0,1,2,3,4,5,6,7,8,9,10,11,12,13,14,15);
        v16bf alo = __builtin_shufflevector(a0l, a1l, 0,1,2,3,4,5,6,7,8,9,10,11,12,13,14,15);
        // B fragment: elems 0..7 -> K = kk+8*half+j ; elems 8..15 -> +16, column n0+l16
        const int kb = kk + half * 8;
        v8bf b0h = *(const v8bf*)(Bhi + bcol + kb);
        v8bf b1h = *(const v8bf*)(Bhi + bcol + kb + 16);
        v8bf b0l = *(const v8bf*)(Blo + bcol + kb);
        v8bf b1l = *(const v8bf*)(Blo + bcol + kb + 16);
        v16bf bhi = __builtin_shufflevector(b0h, b1h, 0,1,2,3,4,5,6,7,8,9,10,11,12,13,14,15);
        v16bf blo = __builtin_shufflevector(b0l, b1l, 0,1,2,3,4,5,6,7,8,9,10,11,12,13,14,15);

        c = __builtin_amdgcn_wmma_f32_16x16x32_bf16(false, ahi, false, bhi, (short)0, c, false, false);
        c = __builtin_amdgcn_wmma_f32_16x16x32_bf16(false, alo, false, bhi, (short)0, c, false, false);
        c = __builtin_amdgcn_wmma_f32_16x16x32_bf16(false, ahi, false, blo, (short)0, c, false, false);
    }
    // D layout: VGPR r -> row m0 + 8*half + r, col n0 + l16
    float* crow = C + (size_t)(m0 + half * 8) * Nn + (n0 + l16);
#pragma unroll
    for (int r = 0; r < 8; ++r) crow[(size_t)r * Nn] = c[r];
}

// ---------------------------------------------------------------------------
// Init per-layer accumulators (amax=-inf encoded, denom=0, out=0)
// ---------------------------------------------------------------------------
__global__ void init_layer_kernel(unsigned int* __restrict__ amax,
                                  float* __restrict__ denom,
                                  float* __restrict__ out,
                                  int N, int H, int HC) {
    int idx = blockIdx.x * blockDim.x + threadIdx.x;
    if (idx < N * H) { amax[idx] = ENC_NEG_INF; denom[idx] = 0.f; }
    if (idx < N * HC) out[idx] = 0.f;
}

// ---------------------------------------------------------------------------
// Edge logits: one wave32 per edge. logit[e,h] = att_h . lrelu(xl[src]+xr[dst])
// Fused atomic segment-max.
// ---------------------------------------------------------------------------
__global__ void edge_logits_kernel(const long long* __restrict__ ei, int E, int Etot,
                                   const float* __restrict__ xl, const float* __restrict__ xr,
                                   const float* __restrict__ att,
                                   float* __restrict__ logits,
                                   unsigned int* __restrict__ amax,
                                   int H, int C) {
    const int wid  = (int)(((size_t)blockIdx.x * blockDim.x + threadIdx.x) >> 5);
    const int lane = threadIdx.x & 31;
    if (wid >= Etot) return;
    int src, dst;
    if (wid < E) { src = (int)ei[wid]; dst = (int)ei[(size_t)E + wid]; }
    else         { src = dst = wid - E; }
    const int HC = H * C;
    const float* pl = xl + (size_t)src * HC;
    const float* pr = xr + (size_t)dst * HC;
    for (int h = 0; h < H; ++h) {
        float partial = 0.f;
        for (int c = lane; c < C; c += 32) {
            float v = pl[h * C + c] + pr[h * C + c];
            v = (v > 0.f) ? v : NEG_SLOPE * v;
            partial += att[h * C + c] * v;
        }
#pragma unroll
        for (int o = 16; o > 0; o >>= 1) partial += __shfl_xor(partial, o, 32);
        if (lane == 0) {
            logits[(size_t)wid * H + h] = partial;
            atomicMax(&amax[(size_t)dst * H + h], enc_f(partial));
        }
    }
}

// ---------------------------------------------------------------------------
// ex = exp(logit - amax[dst]); denom[dst] += ex  (ex stored in-place)
// ---------------------------------------------------------------------------
__global__ void edge_softmax_kernel(const long long* __restrict__ ei, int E, int Etot,
                                    float* __restrict__ logits,
                                    const unsigned int* __restrict__ amax,
                                    float* __restrict__ denom, int H) {
    int idx = blockIdx.x * blockDim.x + threadIdx.x;
    if (idx >= Etot * H) return;
    int e = idx / H, h = idx - e * H;
    int dst = (e < E) ? (int)ei[(size_t)E + e] : e - E;
    float m  = dec_f(amax[(size_t)dst * H + h]);
    float ex = expf(logits[idx] - m);
    logits[idx] = ex;
    atomicAdd(&denom[(size_t)dst * H + h], ex);
}

// ---------------------------------------------------------------------------
// out[dst] += alpha[e,h] * xl[src]   (one wave32 per edge, atomic scatter)
// ---------------------------------------------------------------------------
__global__ void edge_aggregate_kernel(const long long* __restrict__ ei, int E, int Etot,
                                      const float* __restrict__ xl,
                                      const float* __restrict__ logits,
                                      const float* __restrict__ denom,
                                      float* __restrict__ out, int H, int C) {
    const int wid  = (int)(((size_t)blockIdx.x * blockDim.x + threadIdx.x) >> 5);
    const int lane = threadIdx.x & 31;
    if (wid >= Etot) return;
    int src, dst;
    if (wid < E) { src = (int)ei[wid]; dst = (int)ei[(size_t)E + wid]; }
    else         { src = dst = wid - E; }
    const int HC = H * C;
    const float* pl = xl + (size_t)src * HC;
    float* po = out + (size_t)dst * HC;
    for (int c = lane; c < HC; c += 32) {
        int h = c / C;
        float alpha = logits[(size_t)wid * H + h] / denom[(size_t)dst * H + h];
        atomicAdd(&po[c], alpha * pl[c]);
    }
}

// ---------------------------------------------------------------------------
// out = elu(out + bias)
// ---------------------------------------------------------------------------
__global__ void bias_elu_kernel(float* __restrict__ out, const float* __restrict__ bias,
                                int total, int HC) {
    int idx = blockIdx.x * blockDim.x + threadIdx.x;
    if (idx >= total) return;
    float v = out[idx] + bias[idx % HC];
    out[idx] = (v > 0.f) ? v : expm1f(v);
}

// ---------------------------------------------------------------------------
// Host launcher
// ---------------------------------------------------------------------------
extern "C" void kernel_launch(void* const* d_in, const int* in_sizes, int n_in,
                              void* d_out, int out_size, void* d_ws, size_t ws_size,
                              hipStream_t stream) {
    const float*     x    = (const float*)d_in[0];
    const long long* ei   = (const long long*)d_in[1];
    const float*     W1l  = (const float*)d_in[2];
    const float*     W1r  = (const float*)d_in[3];
    const float*     att1 = (const float*)d_in[4];
    const float*     b1   = (const float*)d_in[5];
    const float*     W2l  = (const float*)d_in[6];
    const float*     W2r  = (const float*)d_in[7];
    const float*     att2 = (const float*)d_in[8];
    const float*     b2   = (const float*)d_in[9];
    const float*     W3l  = (const float*)d_in[10];
    const float*     W3r  = (const float*)d_in[11];
    const float*     att3 = (const float*)d_in[12];
    const float*     b3   = (const float*)d_in[13];

    const int N = 50000, E = 800000, Etot = E + N;
    const int K = 128;  // input width of every layer

    // Workspace carving (all offsets 256B-aligned)
    char* ws = (char*)d_ws;
    size_t off = 0;
    auto carve = [&](size_t bytes) -> char* {
        char* p = ws + off;
        off = (off + bytes + 255) & ~(size_t)255;
        return p;
    };
    float*        h_cur  = (float*)       carve((size_t)N * 128 * sizeof(float));
    bf16*         h_hi   = (bf16*)        carve((size_t)N * 128 * sizeof(bf16));
    bf16*         h_lo   = (bf16*)        carve((size_t)N * 128 * sizeof(bf16));
    float*        xl     = (float*)       carve((size_t)N * 128 * sizeof(float));
    float*        xr     = (float*)       carve((size_t)N * 128 * sizeof(float));
    bf16*         wtl_hi = (bf16*)        carve((size_t)128 * 128 * sizeof(bf16));
    bf16*         wtl_lo = (bf16*)        carve((size_t)128 * 128 * sizeof(bf16));
    bf16*         wtr_hi = (bf16*)        carve((size_t)128 * 128 * sizeof(bf16));
    bf16*         wtr_lo = (bf16*)        carve((size_t)128 * 128 * sizeof(bf16));
    float*        logits = (float*)       carve((size_t)Etot * 2 * sizeof(float));
    unsigned int* amax   = (unsigned int*)carve((size_t)N * 2 * sizeof(unsigned int));
    float*        denom  = (float*)       carve((size_t)N * 2 * sizeof(float));

    auto run_layer = [&](const float* hin, const float* Wl, const float* Wr,
                         const float* att, const float* bias,
                         int H, int C, float* out_) {
        const int HC = H * C;
        const int Nn = HC;  // output width of the two GEMMs
        const size_t nfeat = (size_t)N * K;

        // 1) split input features to hi/lo bf16
        split_bf16_kernel<<<(unsigned)((nfeat + 255) / 256), 256, 0, stream>>>(hin, h_hi, h_lo, nfeat);
        // 2) split + transpose weights
        split_w_t_kernel<<<(K * Nn + 255) / 256, 256, 0, stream>>>(Wl, wtl_hi, wtl_lo, K, Nn);
        split_w_t_kernel<<<(K * Nn + 255) / 256, 256, 0, stream>>>(Wr, wtr_hi, wtr_lo, K, Nn);
        // 3) init accumulators (and zero the output, which may alias hin's buffer:
        //    safe because split_bf16 above has already consumed hin on this stream)
        init_layer_kernel<<<(N * HC + 255) / 256, 256, 0, stream>>>(amax, denom, out_, N, H, HC);
        // 4) GEMMs: xl = h @ Wl, xr = h @ Wr  (wave per 16x16 tile, split-bf16 3x wmma)
        dim3 gblk(32 * (Nn / 16));
        wmma_gemm_bf16x3<<<N / 16, gblk, 0, stream>>>(h_hi, h_lo, wtl_hi, wtl_lo, xl, N, K, Nn);
        wmma_gemm_bf16x3<<<N / 16, gblk, 0, stream>>>(h_hi, h_lo, wtr_hi, wtr_lo, xr, N, K, Nn);
        // 5) edge attention logits + segment max
        edge_logits_kernel<<<(unsigned)(((size_t)Etot * 32 + 255) / 256), 256, 0, stream>>>(
            ei, E, Etot, xl, xr, att, logits, amax, H, C);
        // 6) exp + segment sum
        edge_softmax_kernel<<<(Etot * H + 255) / 256, 256, 0, stream>>>(
            ei, E, Etot, logits, amax, denom, H);
        // 7) weighted aggregation (atomic scatter)
        edge_aggregate_kernel<<<(unsigned)(((size_t)Etot * 32 + 255) / 256), 256, 0, stream>>>(
            ei, E, Etot, xl, logits, denom, out_, H, C);
        // 8) bias + ELU
        bias_elu_kernel<<<(N * HC + 255) / 256, 256, 0, stream>>>(out_, bias, N * HC, HC);
    };

    run_layer(x,     W1l, W1r, att1, b1, 2, 64, h_cur);
    run_layer(h_cur, W2l, W2r, att2, b2, 2, 64, h_cur);
    run_layer(h_cur, W3l, W3r, att3, b3, 1, 64, (float*)d_out);

    (void)in_sizes; (void)n_in; (void)out_size; (void)ws_size;
}
